// CausalSelfAttention_5411658793445
// MI455X (gfx1250) — matrix operand order repaired and checked by reading, not compiled, not run on hardware
//
#include <hip/hip_runtime.h>
#include <hip/hip_bf16.h>
#include <stdint.h>

typedef __bf16 bf16_t;
typedef __attribute__((ext_vector_type(16))) __bf16 bf16x16;
typedef __attribute__((ext_vector_type(8)))  __bf16 bf16x8;
typedef __attribute__((ext_vector_type(8)))  float  f32x8;

union BFrag {
  bf16x16 v;
  bf16x8  h[2];
  unsigned short u[16];
};

// Native f32 -> bf16 (clang lowers to HW cvt on gfx1250; avoids manual
// bfe/add3 rounding chains in VALU).
__device__ __forceinline__ unsigned short f2bfu(float f) {
  union { bf16_t b; unsigned short s; } o; o.b = (bf16_t)f; return o.s;
}
__device__ __forceinline__ bf16_t f2bf(float f) { return (bf16_t)f; }

// gfx1250 async global->LDS copy (ASYNCcnt-tracked), GV mode, 16 bytes/lane.
__device__ __forceinline__ void async_b128(uint32_t lds_off, const void* g) {
  asm volatile("global_load_async_to_lds_b128 %0, %1, off"
               :: "v"(lds_off), "v"((uint64_t)(uintptr_t)g)
               : "memory");
}
__device__ __forceinline__ uint32_t lds_off32(const void* p) {
  return (uint32_t)(uintptr_t)p;   // low 32 bits of LDS aperture addr = offset
}

// ---------------------------------------------------------------------------
// GEMM: C[M,N] = A[M,K] @ W[N,K]^T + bias   (f32 in/out, bf16 WMMA compute)
// Block tile 128x128x32, 256 threads = 8 waves (4 M x 2 N), wave tile 32x64.
// ---------------------------------------------------------------------------
#define GBM 128
#define GBN 128
#define GBK 32

__global__ __launch_bounds__(256)
void gemm_rowmajor_wT(const float* __restrict__ A, const float* __restrict__ W,
                      const float* __restrict__ bias, float* __restrict__ C,
                      int M, int N, int K)
{
  __shared__ alignas(16) bf16_t As[GBM][GBK];
  __shared__ alignas(16) bf16_t Bs[GBN][GBK];

  const int tid  = threadIdx.x;
  const int wave = tid >> 5;
  const int lane = tid & 31;
  const int wm   = wave >> 1;
  const int wn   = wave & 1;
  const int bm   = blockIdx.y * GBM;
  const int bn   = blockIdx.x * GBN;
  const int ln16 = lane & 15;
  const int h8   = (lane < 16) ? 0 : 8;
  const int h16  = (lane < 16) ? 0 : 16;

  f32x8 acc[2][4];
  for (int i = 0; i < 2; ++i)
    for (int j = 0; j < 4; ++j)
      for (int r = 0; r < 8; ++r) acc[i][j][r] = 0.0f;

  const int srow = tid >> 1;
  const int scol = (tid & 1) * 16;

  int gm = bm + srow; if (gm > M - 1) gm = M - 1;
  const float* arow = A + (size_t)gm * K + scol;
  const float* wrow = W + (size_t)(bn + srow) * K + scol;

  for (int k0 = 0; k0 < K; k0 += GBK) {
    // ---- stage A tile ----
    {
      const float4* ap = (const float4*)(arow + k0);
      float4 f0 = ap[0], f1 = ap[1], f2 = ap[2], f3 = ap[3];
      union { unsigned short u[16]; uint4 q[2]; } pk;
      pk.u[0]=f2bfu(f0.x);  pk.u[1]=f2bfu(f0.y);  pk.u[2]=f2bfu(f0.z);  pk.u[3]=f2bfu(f0.w);
      pk.u[4]=f2bfu(f1.x);  pk.u[5]=f2bfu(f1.y);  pk.u[6]=f2bfu(f1.z);  pk.u[7]=f2bfu(f1.w);
      pk.u[8]=f2bfu(f2.x);  pk.u[9]=f2bfu(f2.y);  pk.u[10]=f2bfu(f2.z); pk.u[11]=f2bfu(f2.w);
      pk.u[12]=f2bfu(f3.x); pk.u[13]=f2bfu(f3.y); pk.u[14]=f2bfu(f3.z); pk.u[15]=f2bfu(f3.w);
      *(uint4*)&As[srow][scol]     = pk.q[0];
      *(uint4*)&As[srow][scol + 8] = pk.q[1];
    }
    // ---- stage B tile ----
    {
      const float4* wp = (const float4*)(wrow + k0);
      float4 f0 = wp[0], f1 = wp[1], f2 = wp[2], f3 = wp[3];
      union { unsigned short u[16]; uint4 q[2]; } pk;
      pk.u[0]=f2bfu(f0.x);  pk.u[1]=f2bfu(f0.y);  pk.u[2]=f2bfu(f0.z);  pk.u[3]=f2bfu(f0.w);
      pk.u[4]=f2bfu(f1.x);  pk.u[5]=f2bfu(f1.y);  pk.u[6]=f2bfu(f1.z);  pk.u[7]=f2bfu(f1.w);
      pk.u[8]=f2bfu(f2.x);  pk.u[9]=f2bfu(f2.y);  pk.u[10]=f2bfu(f2.z); pk.u[11]=f2bfu(f2.w);
      pk.u[12]=f2bfu(f3.x); pk.u[13]=f2bfu(f3.y); pk.u[14]=f2bfu(f3.z); pk.u[15]=f2bfu(f3.w);
      *(uint4*)&Bs[srow][scol]     = pk.q[0];
      *(uint4*)&Bs[srow][scol + 8] = pk.q[1];
    }
    if (k0 + GBK < K) {                     // prefetch next k-slice into cache
      __builtin_prefetch(arow + k0 + GBK, 0, 1);
      __builtin_prefetch(wrow + k0 + GBK, 0, 1);
    }
    __syncthreads();

    BFrag afrag[2], bfrag[4];
    for (int mt = 0; mt < 2; ++mt) {
      const bf16_t* p = &As[wm * 32 + mt * 16 + ln16][0];
      afrag[mt].h[0] = *(const bf16x8*)(p + h8);
      afrag[mt].h[1] = *(const bf16x8*)(p + 16 + h8);
    }
    for (int nt = 0; nt < 4; ++nt) {
      const bf16_t* p = &Bs[wn * 64 + nt * 16 + ln16][0];
      bfrag[nt].h[0] = *(const bf16x8*)(p + h8);
      bfrag[nt].h[1] = *(const bf16x8*)(p + 16 + h8);
    }
    for (int mt = 0; mt < 2; ++mt)
      for (int nt = 0; nt < 4; ++nt)
        acc[mt][nt] = __builtin_amdgcn_wmma_f32_16x16x32_bf16(
            false, afrag[mt].v, false, bfrag[nt].v,
            (short)0, acc[mt][nt], false, false);
    __syncthreads();
  }

  const int rl = (lane >= 16) ? 8 : 0;
  for (int mt = 0; mt < 2; ++mt)
    for (int nt = 0; nt < 4; ++nt)
      for (int r = 0; r < 8; ++r) {
        int row = bm + wm * 32 + mt * 16 + rl + r;
        int col = bn + wn * 64 + nt * 16 + ln16;
        if (row < M) C[(size_t)row * N + col] = acc[mt][nt][r] + bias[col];
      }
}

// ---------------------------------------------------------------------------
// RMSNorm + 3-axis RoPE (in place). grid.x = token, grid.y = {0:q, 1:k}.
// ---------------------------------------------------------------------------
__global__ __launch_bounds__(128)
void rmsnorm_rope(float* __restrict__ Qb, float* __restrict__ Kb,
                  const float* __restrict__ gq, const float* __restrict__ gk,
                  const float* __restrict__ freqs, const int* __restrict__ curStart)
{
  const int token = blockIdx.x;
  float* buf      = blockIdx.y ? Kb : Qb;
  const float* g  = blockIdx.y ? gk : gq;
  const int tid   = threadIdx.x;

  float v[12];
  const size_t base = (size_t)token * 1536 + (size_t)tid * 12;
  float ss = 0.0f;
  {
    const float4* p = (const float4*)(buf + base);
    float4 a = p[0], b = p[1], c = p[2];
    v[0]=a.x; v[1]=a.y; v[2]=a.z; v[3]=a.w;
    v[4]=b.x; v[5]=b.y; v[6]=b.z; v[7]=b.w;
    v[8]=c.x; v[9]=c.y; v[10]=c.z; v[11]=c.w;
    for (int i = 0; i < 12; ++i) ss += v[i] * v[i];
  }
  __shared__ float red[128];
  red[tid] = ss;
  __syncthreads();
  for (int off = 64; off; off >>= 1) {
    if (tid < off) red[tid] += red[tid + off];
    __syncthreads();
  }
  const float rms = rsqrtf(red[0] * (1.0f / 1536.0f) + 1e-6f);

  const int startFrame = curStart[0] / 880;
  const int f  = token / 880;
  const int rm = token % 880;
  const int hh = rm / 40;
  const int ww = rm % 40;

  for (int p = 0; p < 6; ++p) {
    const int pj = tid * 6 + p;
    const int j  = pj & 63;
    float theta;
    if (j < 22)      theta = freqs[(startFrame + f) * 64 + j];
    else if (j < 43) theta = freqs[hh * 64 + j];
    else             theta = freqs[ww * 64 + j];
    const float c  = __cosf(theta);
    const float sn = __sinf(theta);
    const float x0 = v[2 * p]     * rms * g[tid * 12 + 2 * p];
    const float x1 = v[2 * p + 1] * rms * g[tid * 12 + 2 * p + 1];
    buf[base + 2 * p]     = x0 * c - x1 * sn;
    buf[base + 2 * p + 1] = x0 * sn + x1 * c;
  }
}

// ---------------------------------------------------------------------------
// Pack kept KV window once: Kc[token][1536] bf16 (row-major) and
// Vt[1536][keepStride] bf16 (pre-transposed). Rows >= keep are zero-filled so
// padded async loads never feed NaNs into WMMA. Tile: 64 tokens x 64 dims.
// ---------------------------------------------------------------------------
__global__ __launch_bounds__(256)
void pack_kv(const float* __restrict__ Knew, const float* __restrict__ Vnew,
             const float* __restrict__ Kpri, const float* __restrict__ Vpri,
             bf16_t* __restrict__ Kc, bf16_t* __restrict__ Vt,
             const int* __restrict__ curStart, int S, int priorRows, int keepStride)
{
  const int tid = threadIdx.x;
  const int j0  = blockIdx.x * 64;
  const int d0  = blockIdx.y * 64;
  const int jl  = tid >> 2;
  const int dl  = (tid & 3) * 16;
  const int token = j0 + jl;
  const int d     = d0 + dl;

  const int cs          = curStart[0];
  const int current_end = cs + S;
  const int block_size  = 3 * 880;
  const int block_end   = (cs / block_size + 1) * block_size;
  int keep_from = block_end - 6 * 880; if (keep_from < 0) keep_from = 0;
  int keep = current_end - keep_from;
  const int total = priorRows + S;
  if (keep > total) keep = total;
  const int offs = total - keep;

  unsigned short kk[16], vv[16];
  if (token < keep) {
    const int t = offs + token;
    const float* kp; const float* vp;
    if (t < priorRows) {
      kp = Kpri + (size_t)t * 1536 + d;
      vp = Vpri + (size_t)t * 1536 + d;
    } else {
      kp = Knew + (size_t)(t - priorRows) * 1536 + d;
      vp = Vnew + (size_t)(t - priorRows) * 1536 + d;
    }
    for (int i = 0; i < 16; i += 4) {
      float4 kx = *(const float4*)(kp + i);
      float4 vx = *(const float4*)(vp + i);
      kk[i+0]=f2bfu(kx.x); kk[i+1]=f2bfu(kx.y); kk[i+2]=f2bfu(kx.z); kk[i+3]=f2bfu(kx.w);
      vv[i+0]=f2bfu(vx.x); vv[i+1]=f2bfu(vx.y); vv[i+2]=f2bfu(vx.z); vv[i+3]=f2bfu(vx.w);
    }
  } else {
    for (int i = 0; i < 16; ++i) { kk[i] = 0; vv[i] = 0; }
  }

  // K: same (token, dim) layout, bf16, 32B vector store
  {
    union { unsigned short u[16]; uint4 q[2]; } pk;
    for (int i = 0; i < 16; ++i) pk.u[i] = kk[i];
    uint4* dst = (uint4*)(Kc + (size_t)token * 1536 + d);
    dst[0] = pk.q[0]; dst[1] = pk.q[1];
  }
  // V: transposed scatter (done once, amortized over all M-tiles/heads)
  {
    unsigned short* vt = (unsigned short*)Vt;
    for (int i = 0; i < 16; ++i)
      vt[(size_t)(d + i) * keepStride + token] = vv[i];
  }
}

// ---------------------------------------------------------------------------
// Flash attention over pre-packed bf16 KV. grid = (ceil(S/128), NH).
// 8 waves; wave w owns query rows [16w,16w+16). K/V staged double-buffered
// with GLOBAL_LOAD_ASYNC_TO_LDS_B128: each wave issues 8 async copies for
// chunk s+1, then s_wait_asynccnt 8 (in-order completion => chunk s ready)
// so the DMA of the next chunk overlaps the WMMA work of the current one.
// ---------------------------------------------------------------------------
#define ABM 128
#define BKV 64

__global__ __launch_bounds__(256)
void flash_attn(const float* __restrict__ Q, const bf16_t* __restrict__ Kc,
                const bf16_t* __restrict__ Vt, float* __restrict__ O,
                const int* __restrict__ curStart, int S, int priorRows,
                int keepStride)
{
  __shared__ alignas(16) bf16_t Ks[2][BKV][128];   // [buf][kv][d]   32KB
  __shared__ alignas(16) bf16_t Vs[2][128][BKV];   // [buf][d][kv]   32KB
  __shared__ alignas(16) bf16_t Ps[8][16][BKV];    // per-wave P     16KB

  const int tid  = threadIdx.x;
  const int wave = tid >> 5;
  const int lane = tid & 31;
  const int ln16 = lane & 15;
  const int h8   = (lane < 16) ? 0 : 8;
  const int h16  = (lane < 16) ? 0 : 16;
  const int head = blockIdx.y;
  const int mt0  = blockIdx.x * ABM;

  const int cs          = curStart[0];
  const int current_end = cs + S;
  const int block_size  = 3 * 880;
  const int block_end   = (cs / block_size + 1) * block_size;
  int keep_from = block_end - 6 * 880; if (keep_from < 0) keep_from = 0;
  int keep = current_end - keep_from;
  const int total = priorRows + S;
  if (keep > total) keep = total;

  // ---- Q fragments (A-operand), pre-scaled by 1/sqrt(128) ----
  const float qscale = 0.0883883476483184f;
  int qrow = mt0 + wave * 16 + ln16; if (qrow > S - 1) qrow = S - 1;
  const float* qp = Q + (size_t)qrow * 1536 + head * 128;
  BFrag qf[4];
  for (int c = 0; c < 4; ++c) {
    const float4* p0 = (const float4*)(qp + 32 * c + h8);
    const float4* p1 = (const float4*)(qp + 32 * c + 16 + h8);
    float4 a = p0[0], b = p0[1], cc = p1[0], d = p1[1];
    qf[c].u[0]=f2bfu(a.x*qscale);  qf[c].u[1]=f2bfu(a.y*qscale);
    qf[c].u[2]=f2bfu(a.z*qscale);  qf[c].u[3]=f2bfu(a.w*qscale);
    qf[c].u[4]=f2bfu(b.x*qscale);  qf[c].u[5]=f2bfu(b.y*qscale);
    qf[c].u[6]=f2bfu(b.z*qscale);  qf[c].u[7]=f2bfu(b.w*qscale);
    qf[c].u[8]=f2bfu(cc.x*qscale); qf[c].u[9]=f2bfu(cc.y*qscale);
    qf[c].u[10]=f2bfu(cc.z*qscale);qf[c].u[11]=f2bfu(cc.w*qscale);
    qf[c].u[12]=f2bfu(d.x*qscale); qf[c].u[13]=f2bfu(d.y*qscale);
    qf[c].u[14]=f2bfu(d.z*qscale); qf[c].u[15]=f2bfu(d.w*qscale);
  }

  f32x8 oacc[8];
  for (int n = 0; n < 8; ++n)
    for (int r = 0; r < 8; ++r) oacc[n][r] = 0.0f;
  float mrow[8], lsum[8];
  for (int r = 0; r < 8; ++r) { mrow[r] = -3.0e38f; lsum[r] = 0.0f; }

  // async staging addresses (bf16 elements)
  const int kkv = tid >> 2;                   // 0..63 token row for K stage
  const int kdg = (tid & 3) * 32;             // 32 bf16 = 64B per thread
  const uint32_t ksl0 = lds_off32(&Ks[0][kkv][kdg]);
  const uint32_t ksl1 = lds_off32(&Ks[1][kkv][kdg]);
  const bf16_t* kgb = Kc + (size_t)kkv * 1536 + head * 128 + kdg;

  const int vd  = tid >> 1;                   // 0..127 dim row for V stage
  const int vkg = (tid & 1) * 32;             // 32 bf16 = 64B per thread
  const uint32_t vsl0 = lds_off32(&Vs[0][vd][vkg]);
  const uint32_t vsl1 = lds_off32(&Vs[1][vd][vkg]);
  const bf16_t* vgb = Vt + (size_t)(head * 128 + vd) * keepStride + vkg;

  const int nsteps = (keep + BKV - 1) / BKV;

#define ISSUE_CHUNK(stp, ksl, vsl)                                  \
  do {                                                              \
    const bf16_t* kg = kgb + (size_t)(stp) * BKV * 1536;            \
    const bf16_t* vg = vgb + (stp) * BKV;                           \
    async_b128((ksl),      kg);                                     \
    async_b128((ksl) + 16, kg + 8);                                 \
    async_b128((ksl) + 32, kg + 16);                                \
    async_b128((ksl) + 48, kg + 24);                                \
    async_b128((vsl),      vg);                                     \
    async_b128((vsl) + 16, vg + 8);                                 \
    async_b128((vsl) + 32, vg + 16);                                \
    async_b128((vsl) + 48, vg + 24);                                \
  } while (0)

  ISSUE_CHUNK(0, ksl0, vsl0);

  for (int step = 0; step < nsteps; ++step) {
    const int buf = step & 1;
    if (step + 1 < nsteps) {
      if (buf) ISSUE_CHUNK(step + 1, ksl0, vsl0);
      else     ISSUE_CHUNK(step + 1, ksl1, vsl1);
      // 8 copies in flight for next chunk; in-order completion means
      // asynccnt<=8 guarantees this wave's current-chunk copies landed.
      asm volatile("s_wait_asynccnt 0x8" ::: "memory");
    } else {
      asm volatile("s_wait_asynccnt 0x0" ::: "memory");
    }
    __syncthreads();

    // ---- S = Q @ K^T (16 x 64 per wave) ----
    f32x8 sacc[4];
    for (int t4 = 0; t4 < 4; ++t4)
      for (int r = 0; r < 8; ++r) sacc[t4][r] = 0.0f;
    for (int c = 0; c < 4; ++c) {
      for (int t4 = 0; t4 < 4; ++t4) {
        const bf16_t* p = &Ks[buf][t4 * 16 + ln16][32 * c];
        BFrag bf; bf.h[0] = *(const bf16x8*)(p + h8); bf.h[1] = *(const bf16x8*)(p + 16 + h8);
        sacc[t4] = __builtin_amdgcn_wmma_f32_16x16x32_bf16(
            false, qf[c].v, false, bf.v, (short)0, sacc[t4], false, false);
      }
    }

    // ---- mask columns beyond kept window (partial last chunk) ----
    for (int t4 = 0; t4 < 4; ++t4) {
      const int col = step * BKV + t4 * 16 + ln16;
      if (col >= keep)
        for (int r = 0; r < 8; ++r) sacc[t4][r] = -3.0e38f;
    }

    // ---- online softmax (row stats across 16-lane halves, wave32) ----
    float mnew[8], alpha[8];
    for (int r = 0; r < 8; ++r) {
      float mx = fmaxf(fmaxf(sacc[0][r], sacc[1][r]),
                       fmaxf(sacc[2][r], sacc[3][r]));
      for (int d = 1; d < 16; d <<= 1) mx = fmaxf(mx, __shfl_xor(mx, d, 32));
      mnew[r]  = fmaxf(mrow[r], mx);
      alpha[r] = __expf(mrow[r] - mnew[r]);
    }
    float psum[8];
    for (int r = 0; r < 8; ++r) psum[r] = 0.0f;
    for (int t4 = 0; t4 < 4; ++t4)
      for (int r = 0; r < 8; ++r) {
        const float pv = __expf(sacc[t4][r] - mnew[r]);
        sacc[t4][r] = pv;
        psum[r] += pv;
      }
    for (int r = 0; r < 8; ++r) {
      float sp = psum[r];
      for (int d = 1; d < 16; d <<= 1) sp += __shfl_xor(sp, d, 32);
      lsum[r] = lsum[r] * alpha[r] + sp;
      mrow[r] = mnew[r];
    }
    for (int n = 0; n < 8; ++n)
      for (int r = 0; r < 8; ++r) oacc[n][r] *= alpha[r];

    // ---- P: C-layout -> A-layout via per-wave LDS scratch ----
    {
      const int pr = (lane >= 16) ? 8 : 0;
      for (int t4 = 0; t4 < 4; ++t4)
        for (int r = 0; r < 8; ++r)
          Ps[wave][pr + r][t4 * 16 + ln16] = f2bf(sacc[t4][r]);
    }

    // ---- O += P @ V ----
    for (int kk = 0; kk < 2; ++kk) {
      BFrag pf;
      const bf16_t* pp = &Ps[wave][ln16][kk * 32];
      pf.h[0] = *(const bf16x8*)(pp + h8);
      pf.h[1] = *(const bf16x8*)(pp + 16 + h8);
      for (int n = 0; n < 8; ++n) {
        const bf16_t* vpp = &Vs[buf][n * 16 + ln16][kk * 32];
        BFrag vf; vf.h[0] = *(const bf16x8*)(vpp + h8); vf.h[1] = *(const bf16x8*)(vpp + 16 + h8);
        oacc[n] = __builtin_amdgcn_wmma_f32_16x16x32_bf16(
            false, pf.v, false, vf.v, (short)0, oacc[n], false, false);
      }
    }
    __syncthreads();
  }
#undef ISSUE_CHUNK

  // ---- epilogue: normalize and store (S, NH, HD) f32 ----
  const int rl = (lane >= 16) ? 8 : 0;
  for (int r = 0; r < 8; ++r) {
    const int row = mt0 + wave * 16 + rl + r;
    if (row < S) {
      const float inv = 1.0f / lsum[r];
      for (int n = 0; n < 8; ++n)
        O[(size_t)row * 1536 + head * 128 + n * 16 + ln16] = oacc[n][r] * inv;
    }
  }
}

// ---------------------------------------------------------------------------
extern "C" void kernel_launch(void* const* d_in, const int* in_sizes, int n_in,
                              void* d_out, int out_size, void* d_ws, size_t ws_size,
                              hipStream_t stream)
{
  const float* x     = (const float*)d_in[0];
  const float* freqs = (const float*)d_in[1];
  const float* pk    = (const float*)d_in[2];
  const float* pv    = (const float*)d_in[3];
  const float* Wq    = (const float*)d_in[4];
  const float* bq    = (const float*)d_in[5];
  const float* Wk    = (const float*)d_in[6];
  const float* bk    = (const float*)d_in[7];
  const float* Wv    = (const float*)d_in[8];
  const float* bv    = (const float*)d_in[9];
  const float* Wo    = (const float*)d_in[10];
  const float* bo    = (const float*)d_in[11];
  const float* gq    = (const float*)d_in[12];
  const float* gk    = (const float*)d_in[13];
  const int*   cs    = (const int*)d_in[14];

  const int DIMc = 1536;
  const int S         = in_sizes[0] / DIMc;   // 2640
  const int priorRows = in_sizes[2] / DIMc;   // 5280
  const int keepStride = ((priorRows + S + 63) / 64) * 64;  // padded KV columns

  float* qb = (float*)d_ws;
  float* kb = qb + (size_t)S * DIMc;
  float* vb = kb + (size_t)S * DIMc;
  float* ab = vb + (size_t)S * DIMc;
  bf16_t* Kc = (bf16_t*)(ab + (size_t)S * DIMc);
  bf16_t* Vt = Kc + (size_t)keepStride * DIMc;

  dim3 gg(DIMc / GBN, (S + GBM - 1) / GBM);
  gemm_rowmajor_wT<<<gg, 256, 0, stream>>>(x, Wq, bq, qb, S, DIMc, DIMc);
  gemm_rowmajor_wT<<<gg, 256, 0, stream>>>(x, Wk, bk, kb, S, DIMc, DIMc);
  gemm_rowmajor_wT<<<gg, 256, 0, stream>>>(x, Wv, bv, vb, S, DIMc, DIMc);

  rmsnorm_rope<<<dim3(S, 2), 128, 0, stream>>>(qb, kb, gq, gk, freqs, cs);

  pack_kv<<<dim3(keepStride / 64, DIMc / 64), 256, 0, stream>>>(
      kb, vb, pk, pv, Kc, Vt, cs, S, priorRows, keepStride);

  flash_attn<<<dim3((S + ABM - 1) / ABM, 12), 256, 0, stream>>>(
      qb, Kc, Vt, ab, cs, S, priorRows, keepStride);

  gemm_rowmajor_wT<<<gg, 256, 0, stream>>>(ab, Wo, bo, (float*)d_out, S, DIMc, DIMc);
}
